// SecFedGateClient_48155173323225
// MI455X (gfx1250) — compile-verified, matrix operand order checked by reading, not run on hardware
//
#include <hip/hip_runtime.h>

// ---------------------------------------------------------------------------
// Types for CDNA5 WMMA (wave32): 16x16x32 bf16 -> f32 accumulate
// ---------------------------------------------------------------------------
typedef __bf16 bf16_t;
typedef __attribute__((ext_vector_type(16))) __bf16 bf16x16;
typedef __attribute__((ext_vector_type(8)))  float  f32x8;
typedef __attribute__((ext_vector_type(4)))  unsigned int u32x4;
typedef __attribute__((ext_vector_type(8)))  unsigned int u32x8;

union Frag16 { bf16x16 v; unsigned int w[8]; };

__device__ __forceinline__ bf16_t f2bf(float f) { return (bf16_t)f; }
__device__ __forceinline__ float  bf2f(bf16_t h) { return (float)h; }

// K-offset inside a 16-bit A/B fragment per CDNA5 ISA 7.12.2:
//   VGPR v (0..7), half = lane>>4:  k = (v<4?0:16) + 8*half + 2*(v&3)  (+pair j)
__device__ __forceinline__ int frag_k(int v, int half) {
    return ((v < 4) ? 0 : 16) + 8 * half + 2 * (v & 3);
}

// ---------------------------------------------------------------------------
// Tensor Data Mover: DMA a 2-D bf16 tile (tw x th elements, row stride
// `stride_elems`) from global memory into LDS at byte offset lds_byte_addr.
// Issued once per wave; completion tracked with TENSORcnt (in-order per wave,
// which enables ping-pong double buffering with s_wait_tensorcnt 2).
// D# layout per CDNA5 ISA 8.3/8.4 (group0 128b, group1 256b; groups 2/3 NULL
// for <=2D tiles).
// ---------------------------------------------------------------------------
__device__ __forceinline__ void tdm_load_2d_bf16(unsigned int lds_byte_addr,
                                                 const bf16_t* g,
                                                 unsigned int tw, unsigned int th_,
                                                 unsigned int stride_elems) {
    unsigned long long ga = (unsigned long long)(uintptr_t)g;
    u32x4 g0;
    g0[0] = 1u;                                   // count=1, user-mode, no gather
    g0[1] = lds_byte_addr;                        // lds_addr
    g0[2] = (unsigned int)ga;                     // global_addr[31:0]
    g0[3] = (unsigned int)((ga >> 32) & 0x01FFFFFFu) | 0x80000000u;  // addr[56:32] | type=2
    u32x8 g1;
    g1[0] = 0x00010000u;                          // workgroup_mask=0, data_size=1 (2B)
    g1[1] = (tw & 0xFFFFu) << 16;                 // abar_addr=0 | tensor_dim0[15:0]
    g1[2] = ((tw >> 16) & 0xFFFFu) | ((th_ & 0xFFFFu) << 16);  // td0 hi | tensor_dim1 lo
    g1[3] = ((th_ >> 16) & 0xFFFFu) | ((tw & 0xFFFFu) << 16);  // td1 hi | tile_dim0
    g1[4] = th_ & 0xFFFFu;                        // tile_dim1 | tile_dim2=0
    g1[5] = stride_elems;                         // tensor_dim0_stride[31:0]
    g1[6] = 0u;                                   // stride hi | dim1_stride lo
    g1[7] = 0u;
    asm volatile("tensor_load_to_lds %0, %1, null, null"
                 :: "s"(g0), "s"(g1) : "memory");
}

// ---------------------------------------------------------------------------
// f32 -> bf16 conversion
// ---------------------------------------------------------------------------
__global__ void cvt_f32_to_bf16(const float* __restrict__ src,
                                bf16_t* __restrict__ dst, int n) {
    int i = blockIdx.x * blockDim.x + threadIdx.x;
    if (i < n) dst[i] = f2bf(src[i]);
}

// ---------------------------------------------------------------------------
// adjacency (f32 dense) -> bitmask, 1 bit per edge
// ---------------------------------------------------------------------------
__global__ void adj_to_bits(const float* __restrict__ adj,
                            unsigned* __restrict__ bits, int Nn) {
    int idx = blockIdx.x * blockDim.x + threadIdx.x;
    int words = Nn >> 5;
    if (idx >= Nn * words) return;
    int i = idx / words, wrd = idx - i * words;
    const float* row = adj + (size_t)i * Nn + wrd * 32;
    unsigned m = 0u;
    #pragma unroll
    for (int j = 0; j < 32; ++j)
        if (row[j] > 0.f) m |= (1u << j);
    bits[idx] = m;
}

// ---------------------------------------------------------------------------
// Generic bf16 WMMA GEMM:  C[M,Nc] = act(A[M,K] @ B[K,Nc] + bias (+residual))
// block = 128 threads (4 waves); block computes 16(M) x 64(N); wave w owns the
// n-subtile w*16. The K x 64 B panel is TDM-staged into LDS once per block.
// act: 0 = none, 1 = relu.  Cf / Cb written if non-null.  K <= 256.
// ---------------------------------------------------------------------------
__global__ __launch_bounds__(128)
void gemm_bf16_wmma(const bf16_t* __restrict__ A, const bf16_t* __restrict__ B,
                    const float* __restrict__ bias, const float* __restrict__ residual,
                    float* __restrict__ Cf, bf16_t* __restrict__ Cb,
                    int M, int K, int Nc, int act) {
    __shared__ bf16_t Bs[256 * 64];

    int tid  = threadIdx.x;
    int wave = tid >> 5;
    int lane = tid & 31;
    int half = lane >> 4;
    int l16  = lane & 15;
    int m0 = blockIdx.y * 16;
    int nb = blockIdx.x * 64;

    if (wave == 0) {
        tdm_load_2d_bf16((unsigned int)(uintptr_t)Bs, B + nb,
                         64u, (unsigned int)K, (unsigned int)Nc);
        __builtin_amdgcn_s_wait_tensorcnt(0);
    }
    __syncthreads();

    f32x8 acc = {0.f, 0.f, 0.f, 0.f, 0.f, 0.f, 0.f, 0.f};
    int ncol = wave * 16 + l16;   // column inside the 64-wide panel

    for (int kk = 0; kk < K; kk += 32) {
        Frag16 a, b;
        #pragma unroll
        for (int v = 0; v < 8; ++v) {
            int k0 = kk + frag_k(v, half);
            a.w[v] = *(const unsigned int*)(A + (size_t)(m0 + l16) * K + k0);  // pair (k0,k0+1)
            b.v[2 * v]     = Bs[(k0)     * 64 + ncol];
            b.v[2 * v + 1] = Bs[(k0 + 1) * 64 + ncol];
        }
        acc = __builtin_amdgcn_wmma_f32_16x16x32_bf16(false, a.v, false, b.v,
                                                      (short)0, acc, false, false);
    }

    #pragma unroll
    for (int v = 0; v < 8; ++v) {
        int row = m0 + v + 8 * half;
        int col = nb + wave * 16 + l16;
        float c = acc[v];
        if (bias)     c += bias[col];
        if (residual) c += residual[(size_t)row * Nc + col];
        if (act == 1) c = fmaxf(c, 0.f);
        if (Cf) Cf[(size_t)row * Nc + col] = c;
        if (Cb) Cb[(size_t)row * Nc + col] = f2bf(c);
    }
}

// ---------------------------------------------------------------------------
// clin[i] = sigmoid( dot(hr[i,:], r2_w) + r2_b )
// ---------------------------------------------------------------------------
__global__ void clin_kernel(const bf16_t* __restrict__ hr, const float* __restrict__ r2w,
                            const float* __restrict__ r2b, float* __restrict__ clin,
                            int Nn, int Hh) {
    int i = blockIdx.x * blockDim.x + threadIdx.x;
    if (i >= Nn) return;
    float acc = r2b[0];
    for (int j = 0; j < Hh; ++j) acc += bf2f(hr[(size_t)i * Hh + j]) * r2w[j];
    clin[i] = 1.f / (1.f + __expf(-acc));
}

// ---------------------------------------------------------------------------
// xw = x * sigmoid(gatepre) * clin[row] + x * rel   (writes f32 + bf16 copies)
// ---------------------------------------------------------------------------
__global__ void combine_kernel(const float* __restrict__ x, const float* __restrict__ gatepre,
                               const float* __restrict__ clin, const float* __restrict__ rel,
                               float* __restrict__ xwf, bf16_t* __restrict__ xwb,
                               int Nn, int Dd) {
    int idx = blockIdx.x * blockDim.x + threadIdx.x;
    if (idx >= Nn * Dd) return;
    int i = idx / Dd;
    float g = 1.f / (1.f + __expf(-gatepre[idx]));
    float v = x[idx] * g * clin[i] + x[idx] * rel[idx];
    xwf[idx] = v;
    xwb[idx] = f2bf(v);
}

// ---------------------------------------------------------------------------
// Flash attention, bf16 WMMA core, adjacency bitmask, double-buffered
// TDM-staged K/V tiles (issue tile i+1, s_wait_tensorcnt(2) -> tile i ready).
// grid = (N/64, NH), block = 128 (4 waves). Wave w owns query rows
// [qb+16w, qb+16w+16), full HD=64 output (4 accum tiles).
// ---------------------------------------------------------------------------
__global__ __launch_bounds__(128)
void flash_attn_kernel(const bf16_t* __restrict__ Q, const bf16_t* __restrict__ Km,
                       const bf16_t* __restrict__ Vm, const unsigned* __restrict__ maskbits,
                       bf16_t* __restrict__ attbf, int Nn, int Dd) {
    constexpr int BM = 64, BN = 32, HDI = 64;
    int h   = blockIdx.y;
    int qb  = blockIdx.x * BM;
    int tid = threadIdx.x;
    int wave = tid >> 5, lane = tid & 31, half = lane >> 4, l16 = lane & 15;
    int words = Nn >> 5;

    __shared__ bf16_t  Kt[2][BN * HDI];       // ping-pong [key][hd]
    __shared__ bf16_t  Vt[2][BN * HDI];       // ping-pong [key][hd]
    __shared__ float   Sb[4 * 16 * BN];       // per-wave raw scaled scores
    __shared__ bf16_t  Pb[4 * 16 * BN];       // per-wave softmax probs (bf16)
    __shared__ float   alphaB[4 * 16];
    __shared__ float   lB[4 * 16];
    __shared__ unsigned maskW[BM];

    unsigned int ktb[2] = { (unsigned int)(uintptr_t)&Kt[0][0],
                            (unsigned int)(uintptr_t)&Kt[1][0] };
    unsigned int vtb[2] = { (unsigned int)(uintptr_t)&Vt[0][0],
                            (unsigned int)(uintptr_t)&Vt[1][0] };

    // Q fragments for this wave's 16 query rows (two K-steps over HD=64)
    Frag16 qf[2];
    {
        int qrow = qb + wave * 16 + l16;
        const bf16_t* qp = Q + (size_t)qrow * Dd + h * HDI;
        #pragma unroll
        for (int kk2 = 0; kk2 < 2; ++kk2)
            #pragma unroll
            for (int v = 0; v < 8; ++v)
                qf[kk2].w[v] = *(const unsigned int*)(qp + kk2 * 32 + frag_k(v, half));
    }

    const f32x8 zero8 = {0.f, 0.f, 0.f, 0.f, 0.f, 0.f, 0.f, 0.f};
    f32x8 o[4];
    #pragma unroll
    for (int t = 0; t < 4; ++t) o[t] = zero8;
    float mstate = -3.0e38f, lstate = 0.f;

    // prologue: start DMA of tile 0 into buffer 0
    if (wave == 0) {
        tdm_load_2d_bf16(ktb[0], Km + h * HDI, HDI, BN, Dd);
        tdm_load_2d_bf16(vtb[0], Vm + h * HDI, HDI, BN, Dd);
    }

    int it = 0;
    for (int kb = 0; kb < Nn; kb += BN, ++it) {
        int cur = it & 1;
        __syncthreads();  // all waves done reading buffer cur^1 (tile it-1)

        if (wave == 0 && kb + BN < Nn) {  // stream tile it+1 behind the compute
            tdm_load_2d_bf16(ktb[cur ^ 1], Km + (size_t)(kb + BN) * Dd + h * HDI, HDI, BN, Dd);
            tdm_load_2d_bf16(vtb[cur ^ 1], Vm + (size_t)(kb + BN) * Dd + h * HDI, HDI, BN, Dd);
        }
        if (tid < BM)
            maskW[tid] = maskbits[(size_t)(qb + tid) * words + (kb >> 5)];
        if (wave == 0) {
            // TENSORcnt completes in order: <=2 outstanding means tile `it` landed
            if (kb + BN < Nn) __builtin_amdgcn_s_wait_tensorcnt(2);
            else              __builtin_amdgcn_s_wait_tensorcnt(0);
        }
        __syncthreads();  // tile `it` visible to all waves

        const bf16_t* Ktc = &Kt[cur][0];
        const bf16_t* Vtc = &Vt[cur][0];

        // S = (Q K^T) * 1/sqrt(64): two 16x16 C tiles, two K-steps each
        #pragma unroll
        for (int ct = 0; ct < 2; ++ct) {
            f32x8 s = zero8;
            #pragma unroll
            for (int kk2 = 0; kk2 < 2; ++kk2) {
                Frag16 bf;
                int n = ct * 16 + l16;
                #pragma unroll
                for (int v = 0; v < 8; ++v) {
                    int k0 = kk2 * 32 + frag_k(v, half);
                    bf.w[v] = *(const unsigned int*)&Ktc[n * HDI + k0];  // B[k][n] = K[n][k]
                }
                s = __builtin_amdgcn_wmma_f32_16x16x32_bf16(false, qf[kk2].v, false, bf.v,
                                                            (short)0, s, false, false);
            }
            #pragma unroll
            for (int v = 0; v < 8; ++v) {
                int row = v + 8 * half;
                Sb[(wave * 16 + row) * BN + ct * 16 + l16] = s[v] * 0.125f;
            }
        }
        __syncthreads();

        // online softmax: lanes 0..15 each own one query row of this wave
        if (lane < 16) {
            int r = lane;
            const float* Srow = &Sb[(wave * 16 + r) * BN];
            bf16_t* Prow = &Pb[(wave * 16 + r) * BN];
            unsigned mw = maskW[wave * 16 + r];
            float mold = mstate, mnew = mold;
            #pragma unroll
            for (int j = 0; j < BN; ++j)
                if ((mw >> j) & 1u) mnew = fmaxf(mnew, Srow[j]);
            float rsum = 0.f;
            #pragma unroll
            for (int j = 0; j < BN; ++j) {
                float e = ((mw >> j) & 1u) ? __expf(Srow[j] - mnew) : 0.f;
                Prow[j] = f2bf(e);
                rsum += e;
            }
            float alpha = __expf(mold - mnew);
            lstate = lstate * alpha + rsum;
            mstate = mnew;
            alphaB[wave * 16 + r] = alpha;
        }
        __syncthreads();

        // rescale accumulators, then O += P @ V  (one K-step of 32 keys)
        Frag16 pf;
        #pragma unroll
        for (int v = 0; v < 8; ++v)
            pf.w[v] = *(const unsigned int*)&Pb[(wave * 16 + l16) * BN + frag_k(v, half)];
        #pragma unroll
        for (int nt = 0; nt < 4; ++nt) {
            #pragma unroll
            for (int v = 0; v < 8; ++v)
                o[nt][v] *= alphaB[wave * 16 + v + 8 * half];
            Frag16 vf;
            #pragma unroll
            for (int v = 0; v < 8; ++v) {
                int k0 = frag_k(v, half);
                vf.v[2 * v]     = Vtc[k0 * HDI + nt * 16 + l16];
                vf.v[2 * v + 1] = Vtc[(k0 + 1) * HDI + nt * 16 + l16];
            }
            o[nt] = __builtin_amdgcn_wmma_f32_16x16x32_bf16(false, pf.v, false, vf.v,
                                                            (short)0, o[nt], false, false);
        }
    }

    if (lane < 16) lB[wave * 16 + lane] = fmaxf(lstate, 1e-30f);
    __syncthreads();

    #pragma unroll
    for (int nt = 0; nt < 4; ++nt) {
        #pragma unroll
        for (int v = 0; v < 8; ++v) {
            int row = v + 8 * half;
            float val = o[nt][v] / lB[wave * 16 + row];
            attbf[(size_t)(qb + wave * 16 + row) * Dd + h * HDI + nt * 16 + l16] = f2bf(val);
        }
    }
}

// ---------------------------------------------------------------------------
// LayerNorm over D=256, one block per row, writes bf16
// ---------------------------------------------------------------------------
__global__ __launch_bounds__(256)
void layernorm_kernel(const float* __restrict__ h, const float* __restrict__ g,
                      const float* __restrict__ b, bf16_t* __restrict__ y, int Dd) {
    int row = blockIdx.x;
    int t = threadIdx.x;
    __shared__ float red[256];
    float v = h[(size_t)row * Dd + t];
    red[t] = v;
    __syncthreads();
    for (int s = 128; s > 0; s >>= 1) { if (t < s) red[t] += red[t + s]; __syncthreads(); }
    float mu = red[0] / (float)Dd;
    __syncthreads();
    float d = v - mu;
    red[t] = d * d;
    __syncthreads();
    for (int s = 128; s > 0; s >>= 1) { if (t < s) red[t] += red[t + s]; __syncthreads(); }
    float var = red[0] / (float)Dd;
    y[(size_t)row * Dd + t] = f2bf(d * rsqrtf(var + 1e-5f) * g[t] + b[t]);
}

// ---------------------------------------------------------------------------
// Host orchestration
// ---------------------------------------------------------------------------
extern "C" void kernel_launch(void* const* d_in, const int* in_sizes, int n_in,
                              void* d_out, int out_size, void* d_ws, size_t ws_size,
                              hipStream_t stream) {
    constexpr int N = 4096, D = 256, O = 128, HID = 128, NH = 4;

    const float* x    = (const float*)d_in[0];
    const float* ont  = (const float*)d_in[1];
    const float* adj  = (const float*)d_in[2];
    const float* p1_w = (const float*)d_in[3];  const float* p1_b = (const float*)d_in[4];
    const float* p2_w = (const float*)d_in[5];  const float* p2_b = (const float*)d_in[6];
    const float* g1_w = (const float*)d_in[7];  const float* g1_b = (const float*)d_in[8];
    const float* g2_w = (const float*)d_in[9];  const float* g2_b = (const float*)d_in[10];
    const float* r1_w = (const float*)d_in[11]; const float* r1_b = (const float*)d_in[12];
    const float* r2_w = (const float*)d_in[13]; const float* r2_b = (const float*)d_in[14];
    const float* wq_w = (const float*)d_in[15]; const float* wq_b = (const float*)d_in[16];
    const float* wk_w = (const float*)d_in[17]; const float* wk_b = (const float*)d_in[18];
    const float* wv_w = (const float*)d_in[19]; const float* wv_b = (const float*)d_in[20];
    const float* wo_w = (const float*)d_in[21]; const float* wo_b = (const float*)d_in[22];
    const float* ln_g = (const float*)d_in[23]; const float* ln_b = (const float*)d_in[24];
    const float* l1_w = (const float*)d_in[25]; const float* l1_b = (const float*)d_in[26];
    const float* l2_w = (const float*)d_in[27]; const float* l2_b = (const float*)d_in[28];

    char* wp = (char*)d_ws;
    auto alloc = [&](size_t bytes) -> void* {
        void* p = wp;
        wp += (bytes + 255) & ~(size_t)255;
        return p;
    };

    bf16_t* x_bf   = (bf16_t*)alloc((size_t)N * D * 2);
    bf16_t* ont_bf = (bf16_t*)alloc((size_t)N * O * 2);
    bf16_t* p1b = (bf16_t*)alloc((size_t)O * HID * 2);
    bf16_t* p2b = (bf16_t*)alloc((size_t)HID * D * 2);
    bf16_t* g1b = (bf16_t*)alloc((size_t)D * HID * 2);
    bf16_t* g2b = (bf16_t*)alloc((size_t)HID * D * 2);
    bf16_t* r1b = (bf16_t*)alloc((size_t)O * HID * 2);
    bf16_t* wqb = (bf16_t*)alloc((size_t)D * D * 2);
    bf16_t* wkb = (bf16_t*)alloc((size_t)D * D * 2);
    bf16_t* wvb = (bf16_t*)alloc((size_t)D * D * 2);
    bf16_t* wob = (bf16_t*)alloc((size_t)D * D * 2);
    bf16_t* l1b = (bf16_t*)alloc((size_t)D * 128 * 2);
    bf16_t* l2b = (bf16_t*)alloc((size_t)128 * 64 * 2);
    bf16_t* h1_bf  = (bf16_t*)alloc((size_t)N * HID * 2);
    bf16_t* hr_bf  = (bf16_t*)alloc((size_t)N * HID * 2);
    bf16_t* hg_bf  = (bf16_t*)alloc((size_t)N * HID * 2);
    float*  rel     = (float*)alloc((size_t)N * D * 4);
    float*  gatepre = (float*)alloc((size_t)N * D * 4);
    float*  clin    = (float*)alloc((size_t)N * 4);
    float*  xw_f    = (float*)alloc((size_t)N * D * 4);
    bf16_t* xw_bf   = (bf16_t*)alloc((size_t)N * D * 2);
    bf16_t* q_bf    = (bf16_t*)alloc((size_t)N * D * 2);
    bf16_t* k_bf    = (bf16_t*)alloc((size_t)N * D * 2);
    bf16_t* v_bf    = (bf16_t*)alloc((size_t)N * D * 2);
    unsigned* mbits = (unsigned*)alloc((size_t)N * (N / 32) * 4);
    bf16_t* att_bf  = (bf16_t*)alloc((size_t)N * D * 2);
    float*  h_f     = (float*)alloc((size_t)N * D * 4);
    bf16_t* y_bf    = (bf16_t*)alloc((size_t)N * D * 2);
    bf16_t* m1_bf   = (bf16_t*)alloc((size_t)N * HID * 2);

    auto cvt = [&](const float* s, bf16_t* d, int n) {
        cvt_f32_to_bf16<<<(n + 255) / 256, 256, 0, stream>>>(s, d, n);
    };
    auto gemm = [&](const bf16_t* A, const bf16_t* B, const float* bias, const float* res,
                    float* Cf, bf16_t* Cb, int M, int K, int Nc, int act) {
        dim3 grid(Nc / 64, M / 16);
        gemm_bf16_wmma<<<grid, 128, 0, stream>>>(A, B, bias, res, Cf, Cb, M, K, Nc, act);
    };

    // --- precision conversions ---
    cvt(x, x_bf, N * D);
    cvt(ont, ont_bf, N * O);
    cvt(p1_w, p1b, O * HID);
    cvt(p2_w, p2b, HID * D);
    cvt(g1_w, g1b, D * HID);
    cvt(g2_w, g2b, HID * D);
    cvt(r1_w, r1b, O * HID);
    cvt(wq_w, wqb, D * D);
    cvt(wk_w, wkb, D * D);
    cvt(wv_w, wvb, D * D);
    cvt(wo_w, wob, D * D);
    cvt(l1_w, l1b, D * 128);
    cvt(l2_w, l2b, 128 * 64);

    // --- adjacency bitmask ---
    {
        int tot = N * (N / 32);
        adj_to_bits<<<(tot + 255) / 256, 256, 0, stream>>>(adj, mbits, N);
    }

    // --- CAFW: ontology-guided feature weighting ---
    gemm(ont_bf, p1b, p1_b, nullptr, nullptr, h1_bf, N, O, HID, 1);     // relu(ont@p1)
    gemm(h1_bf, p2b, p2_b, nullptr, rel, nullptr, N, HID, D, 0);        // rel
    gemm(ont_bf, r1b, r1_b, nullptr, nullptr, hr_bf, N, O, HID, 1);     // relu(ont@r1)
    clin_kernel<<<(N + 255) / 256, 256, 0, stream>>>(hr_bf, r2_w, r2_b, clin, N, HID);
    gemm(x_bf, g1b, g1_b, nullptr, nullptr, hg_bf, N, D, HID, 1);       // relu(x@g1)
    gemm(hg_bf, g2b, g2_b, nullptr, gatepre, nullptr, N, HID, D, 0);    // gate pre-sigmoid
    combine_kernel<<<(N * D + 255) / 256, 256, 0, stream>>>(x, gatepre, clin, rel,
                                                            xw_f, xw_bf, N, D);

    // --- attention: QKV projections ---
    gemm(xw_bf, wqb, wq_b, nullptr, nullptr, q_bf, N, D, D, 0);
    gemm(xw_bf, wkb, wk_b, nullptr, nullptr, k_bf, N, D, D, 0);
    gemm(xw_bf, wvb, wv_b, nullptr, nullptr, v_bf, N, D, D, 0);

    // --- flash attention with adjacency mask (double-buffered TDM) ---
    {
        dim3 grid(N / 64, NH);
        flash_attn_kernel<<<grid, 128, 0, stream>>>(q_bf, k_bf, v_bf, mbits, att_bf, N, D);
    }

    // --- output projection + residual, layernorm ---
    gemm(att_bf, wob, wo_b, xw_f, h_f, nullptr, N, D, D, 0);            // h = xw + att@wo
    layernorm_kernel<<<N, 256, 0, stream>>>(h_f, ln_g, ln_b, y_bf, D);

    // --- local model head ---
    gemm(y_bf, l1b, l1_b, nullptr, nullptr, m1_bf, N, D, 128, 1);       // relu(y@l1)
    gemm(m1_bf, l2b, l2_b, nullptr, (float*)d_out, nullptr, N, 128, 64, 0);
}
